// EdgeUpdate_38311108280938
// MI455X (gfx1250) — matrix operand order repaired and checked by reading, not compiled, not run on hardware
//
#include <hip/hip_runtime.h>
#include <hip/hip_bf16.h>

#define NNODES 10000
#define NEDGES 640000
#define DNODE  64
#define DEDGE  128
#define NTILES (NEDGES / 16)   // 40000, exact

typedef __attribute__((ext_vector_type(16))) __bf16 v16bf;
typedef __attribute__((ext_vector_type(8)))  float  v8f;

// sched_barrier masks (LLVM amdgcn):
//  0x01 = non-mem ALU may cross, 0x20 = VMEM reads may cross.
//  DS (0x80/0x100) and WMMA (0x08) stay pinned in all cases below.
#define SB_ALU_VMEMRD 0x21
#define SB_ALU_ONLY   0x01

__device__ __forceinline__ unsigned short f2bf_u(float x) {
    // RNE fp32->bf16 (prep kernel only; runs once over 48K elems)
    unsigned u = __float_as_uint(x);
    return (unsigned short)((u + 0x7FFFu + ((u >> 16) & 1u)) >> 16);
}

__device__ __forceinline__ float silu_f(float x) {
    return x / (1.0f + __expf(-x));
}

// native casts -> v_cvt_pk_bf16_f32 pairs
__device__ __forceinline__ v16bf cvt16(float4 a, float4 b, float4 c, float4 d) {
    v16bf r;
    r[0]  = (__bf16)a.x; r[1]  = (__bf16)a.y; r[2]  = (__bf16)a.z; r[3]  = (__bf16)a.w;
    r[4]  = (__bf16)b.x; r[5]  = (__bf16)b.y; r[6]  = (__bf16)b.z; r[7]  = (__bf16)b.w;
    r[8]  = (__bf16)c.x; r[9]  = (__bf16)c.y; r[10] = (__bf16)c.z; r[11] = (__bf16)c.w;
    r[12] = (__bf16)d.x; r[13] = (__bf16)d.y; r[14] = (__bf16)d.z; r[15] = (__bf16)d.w;
    return r;
}

// ---------------------------------------------------------------------------
// Pack W1^T / W2^T (fp32, [in,out]) into bf16 WMMA A-fragments.
// 16-bit A-matrix 16x32 k order used here (contiguous, not the ISA table order):
//   lane<16 : row M=lane,    elem q -> K = q
//   lane>=16: row M=lane-16, elem q -> K = q+16
// A = W^T: A[M][K] = W[32k+K][16m+M].
// w1f: [8 m][8 k][32 lane][16 elem], w2f: [8 m][4 k][32 lane][16 elem]
// ---------------------------------------------------------------------------
__global__ void pack_weights_kernel(const float* __restrict__ W1,
                                    const float* __restrict__ W2,
                                    unsigned short* __restrict__ w1f,
                                    unsigned short* __restrict__ w2f) {
    int t = blockIdx.x * blockDim.x + threadIdx.x;
    if (t < 8 * 8 * 32 * 16) {
        int idx  = t & 15;
        int lane = (t >> 4) & 31;
        int k    = (t >> 9) & 7;
        int m    = (t >> 12) & 7;
        int koff = 16 * (lane >> 4) + idx;
        w1f[t] = f2bf_u(W1[(32 * k + koff) * DEDGE + 16 * m + (lane & 15)]);
    }
    if (t < 8 * 4 * 32 * 16) {
        int idx  = t & 15;
        int lane = (t >> 4) & 31;
        int k    = (t >> 9) & 3;
        int m    = (t >> 11) & 7;
        int koff = 16 * (lane >> 4) + idx;
        w2f[t] = f2bf_u(W2[(32 * k + koff) * DEDGE + 16 * m + (lane & 15)]);
    }
}

// ---------------------------------------------------------------------------
// Fused kernel: one wave handles 16 edges end-to-end.
// ---------------------------------------------------------------------------
__global__ void __launch_bounds__(256)
__attribute__((amdgpu_waves_per_eu(4)))
edge_mlp_kernel(const float* __restrict__ node_scalars,
                const float* __restrict__ edge_feats,
                const int*   __restrict__ src_idxs,
                const int*   __restrict__ dst_idxs,
                const unsigned short* __restrict__ w1f,
                const unsigned short* __restrict__ w2f,
                const float* __restrict__ b1,
                const float* __restrict__ b2,
                const float* __restrict__ gamma,
                const float* __restrict__ beta,
                float* __restrict__ out) {
    extern __shared__ __align__(16) unsigned char smem[];

    // Stage W1 fragments (64 KB) into LDS once per block.
    {
        uint4* s = (uint4*)smem;
        const uint4* g = (const uint4*)w1f;
        const int n16 = (8 * 8 * 32 * 16 * 2) / 16;  // 4096 uint4
        for (int i = threadIdx.x; i < n16; i += blockDim.x) s[i] = g[i];
    }
    __syncthreads();

    const v16bf* A1 = (const v16bf*)smem;   // W1^T fragments in LDS
    const v16bf* A2 = (const v16bf*)w2f;    // W2^T fragments, L2-resident

    const int lane = threadIdx.x & 31;
    const int hl   = lane >> 4;             // half-wave select
    const int wave  = (blockIdx.x * blockDim.x + threadIdx.x) >> 5;
    const int nwave = (gridDim.x * blockDim.x) >> 5;

    for (int tile = wave; tile < NTILES; tile += nwave) {
        const int e   = tile * 16 + (lane & 15);
        const int src = src_idxs[e];
        const int dst = dst_idxs[e];
        const float* srow = node_scalars + src * DNODE;
        const float* drow = node_scalars + dst * DNODE;
        const float* erow = edge_feats + (long)e * DEDGE;

        // -------- Layer 1: h1^T = W1^T x x^T --------
        v8f c1[8] = {};
        #pragma unroll
        for (int k = 0; k < 8; ++k) {
            // 16 contiguous input features per lane starting at 32k + 16*hl;
            // each 16-chunk lies entirely in one of {src,dst,edge} regions.
            const float* base = (k < 2) ? srow : ((k < 4) ? drow : erow);
            const int rb      = (k < 2) ? 0 : ((k < 4) ? 2 : 4);
            const float4* p4  = (const float4*)(base + 32 * (k - rb) + 16 * hl);
            float4 v0 = p4[0], v1 = p4[1], v2 = p4[2], v3 = p4[3];
            v16bf bfrag = cvt16(v0, v1, v2, v3);

            v16bf a[8];
            #pragma unroll
            for (int m = 0; m < 8; ++m) a[m] = A1[(m * 8 + k) * 32 + lane];
            #pragma unroll
            for (int m = 0; m < 8; ++m)
                c1[m] = __builtin_amdgcn_wmma_f32_16x16x32_bf16(
                    false, a[m], false, bfrag, (short)0, c1[m], false, false);
            // pin DS+WMMA inside this k-group; let ALU + VMEM reads (next k's
            // activation loads) schedule freely across.
            __builtin_amdgcn_sched_barrier(SB_ALU_VMEMRD);
        }

        // bias + SiLU (C layout: lane holds edge e, features j = 16m + 8*hl + v)
        #pragma unroll
        for (int m = 0; m < 8; ++m) {
            const int j0 = 16 * m + 8 * hl;
            float4 bb0 = ((const float4*)(b1 + j0))[0];
            float4 bb1 = ((const float4*)(b1 + j0))[1];
            float bb[8] = {bb0.x, bb0.y, bb0.z, bb0.w, bb1.x, bb1.y, bb1.z, bb1.w};
            #pragma unroll
            for (int v = 0; v < 8; ++v) c1[m][v] = silu_f(c1[m][v] + bb[v]);
        }

        // -------- In-register transpose C1 -> layer-2 B fragments --------
        v16bf b2f[4];
        #pragma unroll
        for (int t = 0; t < 4; ++t) {
            #pragma unroll
            for (int v = 0; v < 8; ++v) {
                float a0 = c1[2 * t][v];
                float a1 = c1[2 * t + 1][v];
                float o0 = __shfl_xor(a0, 16, 32);
                float o1 = __shfl_xor(a1, 16, 32);
                float lo = hl ? o1 : a0;   // K elems 0..7 of this half
                float hi = hl ? a1 : o0;   // K elems 8..15
                b2f[t][v]     = (__bf16)lo;
                b2f[t][v + 8] = (__bf16)hi;
            }
        }

        // -------- Layer 2: h2^T = W2^T x h1^T --------
        v8f c2[8] = {};
        #pragma unroll
        for (int m = 0; m < 8; ++m) {
            v16bf a[4];
            #pragma unroll
            for (int t = 0; t < 4; ++t) a[t] = A2[(m * 4 + t) * 32 + lane];
            #pragma unroll
            for (int t = 0; t < 4; ++t)
                c2[m] = __builtin_amdgcn_wmma_f32_16x16x32_bf16(
                    false, a[t], false, b2f[t], (short)0, c2[m], false, false);
            // strict: keep A2 global loads inside their m-group (else the
            // scheduler hoists 32 fragments = 256 VGPRs).
            __builtin_amdgcn_sched_barrier(SB_ALU_ONLY);
        }

        // -------- bias + SiLU + residual --------
        float ssum = 0.0f;
        #pragma unroll
        for (int m = 0; m < 8; ++m) {
            const int j0 = 16 * m + 8 * hl;
            float4 bb0 = ((const float4*)(b2 + j0))[0];
            float4 bb1 = ((const float4*)(b2 + j0))[1];
            float4 e0  = ((const float4*)(erow + j0))[0];
            float4 e1  = ((const float4*)(erow + j0))[1];
            float bb[8] = {bb0.x, bb0.y, bb0.z, bb0.w, bb1.x, bb1.y, bb1.z, bb1.w};
            float ef[8] = {e0.x, e0.y, e0.z, e0.w, e1.x, e1.y, e1.z, e1.w};
            #pragma unroll
            for (int v = 0; v < 8; ++v) {
                float x = ef[v] + silu_f(c2[m][v] + bb[v]);
                c2[m][v] = x;
                ssum += x;
            }
            __builtin_amdgcn_sched_barrier(SB_ALU_ONLY);
        }

        // -------- LayerNorm over 128 (features live in lanes e and e+16) ----
        ssum += __shfl_xor(ssum, 16, 32);
        const float mean = ssum * (1.0f / 128.0f);

        float vsum = 0.0f;
        #pragma unroll
        for (int m = 0; m < 8; ++m) {
            #pragma unroll
            for (int v = 0; v < 8; ++v) {
                float d = c2[m][v] - mean;
                vsum += d * d;
            }
        }
        vsum += __shfl_xor(vsum, 16, 32);
        const float rstd = rsqrtf(vsum * (1.0f / 128.0f) + 1e-5f);

        float* orow = out + (long)e * DEDGE;
        #pragma unroll
        for (int m = 0; m < 8; ++m) {
            const int j0 = 16 * m + 8 * hl;
            float4 g0 = ((const float4*)(gamma + j0))[0];
            float4 g1 = ((const float4*)(gamma + j0))[1];
            float4 t0 = ((const float4*)(beta + j0))[0];
            float4 t1 = ((const float4*)(beta + j0))[1];
            float4 o0, o1;
            o0.x = (c2[m][0] - mean) * rstd * g0.x + t0.x;
            o0.y = (c2[m][1] - mean) * rstd * g0.y + t0.y;
            o0.z = (c2[m][2] - mean) * rstd * g0.z + t0.z;
            o0.w = (c2[m][3] - mean) * rstd * g0.w + t0.w;
            o1.x = (c2[m][4] - mean) * rstd * g1.x + t1.x;
            o1.y = (c2[m][5] - mean) * rstd * g1.y + t1.y;
            o1.z = (c2[m][6] - mean) * rstd * g1.z + t1.z;
            o1.w = (c2[m][7] - mean) * rstd * g1.w + t1.w;
            ((float4*)(orow + j0))[0] = o0;
            ((float4*)(orow + j0))[1] = o1;
            __builtin_amdgcn_sched_barrier(SB_ALU_ONLY);
        }
    }
}

extern "C" void kernel_launch(void* const* d_in, const int* in_sizes, int n_in,
                              void* d_out, int out_size, void* d_ws, size_t ws_size,
                              hipStream_t stream) {
    const float* node_scalars = (const float*)d_in[0];
    const float* edge_feats   = (const float*)d_in[1];
    const int*   src_idxs     = (const int*)d_in[2];
    const int*   dst_idxs     = (const int*)d_in[3];
    const float* W1           = (const float*)d_in[4];
    const float* b1           = (const float*)d_in[5];
    const float* W2           = (const float*)d_in[6];
    const float* b2           = (const float*)d_in[7];
    const float* gamma        = (const float*)d_in[8];
    const float* beta         = (const float*)d_in[9];
    float* out = (float*)d_out;

    unsigned short* w1f = (unsigned short*)d_ws;                                 // 64 KB
    unsigned short* w2f = (unsigned short*)((char*)d_ws + 8 * 8 * 32 * 16 * 2);  // +32 KB

    pack_weights_kernel<<<128, 256, 0, stream>>>(W1, W2, w1f, w2f);

    edge_mlp_kernel<<<1280, 256, 64 * 1024, stream>>>(
        node_scalars, edge_feats, src_idxs, dst_idxs,
        w1f, w2f, b1, b2, gamma, beta, out);
}
